// SentimentLSTM_58600533787457
// MI455X (gfx1250) — matrix operand order repaired and checked by reading, not compiled, not run on hardware
//
#include <hip/hip_runtime.h>
#include <hip/hip_bf16.h>

typedef _Float16 v16h __attribute__((ext_vector_type(16)));
typedef _Float16 h8   __attribute__((ext_vector_type(8)));
typedef float    v8f  __attribute__((ext_vector_type(8)));

#define B_  256
#define T_  512
#define E_  300
#define H_  256
#define FH  1024   // 4*H
#define KP  320    // E padded to multiple of 32
#define BT  (B_ * T_)

// ---------------------------------------------------------------------------
// WMMA fragment loaders (contiguous K order used by this file, both operands K-major in memory)
// A fragment 16x32 f16: lane L holds row (L&15), K = (L>>4)*16 + {0..15}
// ---------------------------------------------------------------------------
__device__ __forceinline__ v16h load_a_frag(const _Float16* A, int ld, int m0,
                                            int k0, int lane) {
  int r  = lane & 15;
  int kb = (lane >> 4) * 16;
  const _Float16* p = A + (size_t)(m0 + r) * ld + k0 + kb;
  return *(const v16h*)(p);
}

// B fragment 32x16 f16 with B stored as [N,K] row-major:
// lane L holds N-row (L&15), K = (L>>4)*16 + {0..15} (contiguous)
__device__ __forceinline__ v16h load_b_frag(const _Float16* Bm, int ld, int n0,
                                            int k0, int lane) {
  int n  = lane & 15;
  int kb = (lane >> 4) * 16;
  return *(const v16h*)(Bm + (size_t)(n0 + n) * ld + k0 + kb);
}

// ---------------------------------------------------------------------------
// 1) Embedding gather + fp16 convert, K padded 300 -> 320 with zeros
// ---------------------------------------------------------------------------
__global__ void gather_embed(const int* __restrict__ x,
                             const float* __restrict__ emb,
                             _Float16* __restrict__ embH) {
  size_t tid = (size_t)blockIdx.x * blockDim.x + threadIdx.x;
  if (tid >= (size_t)BT * KP) return;
  int row = (int)(tid / KP);
  int col = (int)(tid - (size_t)row * KP);
  float v = 0.f;
  if (col < E_) {
    int tok = x[row];
    v = emb[(size_t)tok * E_ + col];
  }
  embH[tid] = (_Float16)v;
}

// ---------------------------------------------------------------------------
// 2) Weight conversion: w_ih padded [2][FH][KP] f16, w_hh [2][FH][H] f16,
//    bias = b_ih + b_hh as fp32 [2][FH]
// ---------------------------------------------------------------------------
__global__ void convert_weights(const float* wihF, const float* whhF,
                                const float* bihF, const float* bhhF,
                                const float* wihB, const float* whhB,
                                const float* bihB, const float* bhhB,
                                _Float16* wihH, _Float16* whhH, float* biasF) {
  int tid = blockIdx.x * blockDim.x + threadIdx.x;
  const int N1 = 2 * FH * KP;
  const int N2 = 2 * FH * H_;
  const int N3 = 2 * FH;
  if (tid < N1) {
    int dir = tid / (FH * KP);
    int rem = tid % (FH * KP);
    int g = rem / KP, col = rem % KP;
    const float* w = dir ? wihB : wihF;
    wihH[tid] = (_Float16)(col < E_ ? w[g * E_ + col] : 0.f);
  } else if (tid < N1 + N2) {
    int u = tid - N1;
    int dir = u / (FH * H_);
    int rem = u % (FH * H_);
    whhH[u] = (_Float16)((dir ? whhB : whhF)[rem]);
  } else if (tid < N1 + N2 + N3) {
    int u = tid - N1 - N2;
    int dir = u / FH, g = u % FH;
    biasF[u] = dir ? (bihB[g] + bhhB[g]) : (bihF[g] + bhhF[g]);
  }
}

// ---------------------------------------------------------------------------
// 3) Zero-init recurrent state (re-done every call: deterministic)
//    h16 is double-buffered: [2 buf][2 dir][B][H]
// ---------------------------------------------------------------------------
__global__ void init_state(_Float16* h16, float* h32, float* c32) {
  int tid = blockIdx.x * blockDim.x + threadIdx.x;
  if (tid < 2 * B_ * H_) {
    h16[tid] = (_Float16)0.f;
    h16[2 * B_ * H_ + tid] = (_Float16)0.f;
    h32[tid] = 0.f;
    c32[tid] = 0.f;
  }
}

// ---------------------------------------------------------------------------
// 4) Input projection GEMM: xp[dir][bt][g] = embH[bt,:] . wih[dir][g,:] + bias
//    Block tile 128x128, 8 waves (4M x 2N), each wave 2x4 WMMA accumulators.
// ---------------------------------------------------------------------------
__global__ void __launch_bounds__(256)
proj_gemm(const _Float16* __restrict__ embH, const _Float16* __restrict__ wihH,
          const float* __restrict__ biasF, _Float16* __restrict__ xpH) {
  int dir = blockIdx.z;
  const _Float16* A  = embH;                                 // [BT, KP]
  const _Float16* Bm = wihH + (size_t)dir * FH * KP;         // [FH, KP]
  const float* bias  = biasF + dir * FH;
  _Float16* out      = xpH + (size_t)dir * BT * FH;

  int lane = threadIdx.x & 31;
  int wid  = threadIdx.x >> 5;
  int m0 = blockIdx.y * 128 + (wid & 3) * 32;
  int n0 = blockIdx.x * 128 + (wid >> 2) * 64;

  v8f zero = {};
  v8f acc[2][4];
#pragma unroll
  for (int i = 0; i < 2; ++i)
#pragma unroll
    for (int j = 0; j < 4; ++j) acc[i][j] = zero;

  for (int k0 = 0; k0 < KP; k0 += 32) {
    if (k0 + 32 < KP)
      __builtin_prefetch(A + (size_t)(m0 + (lane & 15)) * KP + k0 + 32, 0, 3);
    v16h a0 = load_a_frag(A, KP, m0,      k0, lane);
    v16h a1 = load_a_frag(A, KP, m0 + 16, k0, lane);
#pragma unroll
    for (int j = 0; j < 4; ++j) {
      v16h b = load_b_frag(Bm, KP, n0 + 16 * j, k0, lane);
      acc[0][j] = __builtin_amdgcn_wmma_f32_16x16x32_f16(
          false, a0, false, b, (short)0, acc[0][j], false, false);
      acc[1][j] = __builtin_amdgcn_wmma_f32_16x16x32_f16(
          false, a1, false, b, (short)0, acc[1][j], false, false);
    }
  }

  int n  = lane & 15;
  int mo = (lane >> 4) * 8;
#pragma unroll
  for (int i = 0; i < 2; ++i)
#pragma unroll
    for (int j = 0; j < 4; ++j) {
      int col = n0 + 16 * j + n;
      float bc = bias[col];
#pragma unroll
      for (int e = 0; e < 8; ++e) {
        int row = m0 + 16 * i + mo + e;
        out[(size_t)row * FH + col] = (_Float16)(acc[i][j][e] + bc);
      }
    }
}

// ---------------------------------------------------------------------------
// 5) Fused per-timestep kernel: recurrent WMMA GEMM + LSTM cell.
//    Gate-aligned tiling: each wave computes the SAME 16 hidden-unit columns
//    for all 4 gates (N = gate*H + j0), so acc[i][0..3][e] are (i,f,g,o) of
//    one (b,j) pair and the cell update is pure in-register math.
//    h is ping-pong buffered across steps (hprev read, hnext written).
// ---------------------------------------------------------------------------
__global__ void __launch_bounds__(256)
step_fused(const _Float16* __restrict__ hprev, _Float16* __restrict__ hnext,
           const _Float16* __restrict__ whhH, const _Float16* __restrict__ xpH,
           const int* __restrict__ lengths, float* __restrict__ c32,
           float* __restrict__ h32, int step) {
  int dir = blockIdx.z;
  const _Float16* A  = hprev + (size_t)dir * B_ * H_;   // [B, H]
  const _Float16* Bm = whhH + (size_t)dir * FH * H_;    // [FH, H]
  const _Float16* xp = xpH + (size_t)dir * BT * FH;
  int t = dir ? (T_ - 1 - step) : step;

  int lane = threadIdx.x & 31;
  int wid  = threadIdx.x >> 5;
  int m0 = blockIdx.y * 128 + (wid & 3) * 32;      // batch-row base
  int jb = blockIdx.x * 32 + (wid >> 2) * 16;      // hidden-unit base

  v8f zero = {};
  v8f acc[2][4];                                    // [m-frag][gate]
#pragma unroll
  for (int i = 0; i < 2; ++i)
#pragma unroll
    for (int g = 0; g < 4; ++g) acc[i][g] = zero;

#pragma unroll
  for (int k0 = 0; k0 < H_; k0 += 32) {
    v16h a0 = load_a_frag(A, H_, m0,      k0, lane);
    v16h a1 = load_a_frag(A, H_, m0 + 16, k0, lane);
#pragma unroll
    for (int g = 0; g < 4; ++g) {
      v16h b = load_b_frag(Bm, H_, g * H_ + jb, k0, lane);
      acc[0][g] = __builtin_amdgcn_wmma_f32_16x16x32_f16(
          false, a0, false, b, (short)0, acc[0][g], false, false);
      acc[1][g] = __builtin_amdgcn_wmma_f32_16x16x32_f16(
          false, a1, false, b, (short)0, acc[1][g], false, false);
    }
  }

  // Fused LSTM cell epilogue (after all WMMAs; EXEC fully set above).
  int j  = jb + (lane & 15);
  int mo = (lane >> 4) * 8;
  size_t sbase = (size_t)dir * B_ * H_;             // state base for this dir
#pragma unroll
  for (int i = 0; i < 2; ++i) {
#pragma unroll
    for (int e = 0; e < 8; ++e) {
      int b = m0 + 16 * i + mo + e;                 // batch index
      size_t xrow = ((size_t)b * T_ + t) * FH;
      float gi = acc[i][0][e] + (float)xp[xrow + 0 * H_ + j];
      float gf = acc[i][1][e] + (float)xp[xrow + 1 * H_ + j];
      float gg = acc[i][2][e] + (float)xp[xrow + 2 * H_ + j];
      float go = acc[i][3][e] + (float)xp[xrow + 3 * H_ + j];
      size_t idx = sbase + (size_t)b * H_ + j;
      if (t < lengths[b]) {
        float i_ = 1.f / (1.f + __expf(-gi));
        float f_ = 1.f / (1.f + __expf(-gf));
        float g_ = tanhf(gg);
        float o_ = 1.f / (1.f + __expf(-go));
        float c  = f_ * c32[idx] + i_ * g_;
        float h  = o_ * tanhf(c);
        c32[idx] = c;
        h32[idx] = h;
        hnext[idx] = (_Float16)h;
      } else {
        hnext[idx] = hprev[idx];                    // carry frozen state
      }
    }
  }
}

// ---------------------------------------------------------------------------
// 6) Final FC: out[b] = [h_f ; h_b] . fc_w + fc_b
// ---------------------------------------------------------------------------
__global__ void final_fc(const float* __restrict__ h32,
                         const float* __restrict__ fc_w,
                         const float* __restrict__ fc_b,
                         float* __restrict__ out) {
  int b = threadIdx.x;
  float acc = fc_b[0];
  for (int j = 0; j < H_; ++j) {
    acc += h32[b * H_ + j] * fc_w[j];                    // forward half
    acc += h32[B_ * H_ + b * H_ + j] * fc_w[H_ + j];     // backward half
  }
  out[b] = acc;
}

// ---------------------------------------------------------------------------
extern "C" void kernel_launch(void* const* d_in, const int* in_sizes, int n_in,
                              void* d_out, int out_size, void* d_ws,
                              size_t ws_size, hipStream_t stream) {
  (void)in_sizes; (void)n_in; (void)out_size; (void)ws_size;
  const int*   x    = (const int*)d_in[0];
  const int*   len  = (const int*)d_in[1];
  const float* emb  = (const float*)d_in[2];
  const float* wihF = (const float*)d_in[3];
  const float* whhF = (const float*)d_in[4];
  const float* bihF = (const float*)d_in[5];
  const float* bhhF = (const float*)d_in[6];
  const float* wihB = (const float*)d_in[7];
  const float* whhB = (const float*)d_in[8];
  const float* bihB = (const float*)d_in[9];
  const float* bhhB = (const float*)d_in[10];
  const float* fc_w = (const float*)d_in[11];
  const float* fc_b = (const float*)d_in[12];
  float* out = (float*)d_out;

  char* ws = (char*)d_ws;
  size_t off = 0;
  auto carve = [&](size_t bytes) {
    char* p = ws + off;
    off += (bytes + 255) & ~(size_t)255;
    return p;
  };
  _Float16* embH  = (_Float16*)carve((size_t)BT * KP * 2);          //  84 MB
  _Float16* xpH   = (_Float16*)carve((size_t)2 * BT * FH * 2);      // 537 MB
  _Float16* wihH  = (_Float16*)carve((size_t)2 * FH * KP * 2);
  _Float16* whhH  = (_Float16*)carve((size_t)2 * FH * H_ * 2);
  float*    biasF = (float*)carve((size_t)2 * FH * 4);
  _Float16* h16   = (_Float16*)carve((size_t)2 * 2 * B_ * H_ * 2);  // ping-pong
  float*    h32   = (float*)carve((size_t)2 * B_ * H_ * 4);
  float*    c32   = (float*)carve((size_t)2 * B_ * H_ * 4);

  gather_embed<<<(unsigned)(((size_t)BT * KP + 255) / 256), 256, 0, stream>>>(
      x, emb, embH);

  int wc_total = 2 * FH * KP + 2 * FH * H_ + 2 * FH;
  convert_weights<<<(wc_total + 255) / 256, 256, 0, stream>>>(
      wihF, whhF, bihF, bhhF, wihB, whhB, bihB, bhhB, wihH, whhH, biasF);

  init_state<<<(2 * B_ * H_ + 255) / 256, 256, 0, stream>>>(h16, h32, c32);

  proj_gemm<<<dim3(FH / 128, BT / 128, 2), 256, 0, stream>>>(embH, wihH,
                                                             biasF, xpH);

  const size_t HBUF = (size_t)2 * B_ * H_;   // one ping-pong buffer (both dirs)
  for (int s = 0; s < T_; ++s) {
    _Float16* hprev = h16 + (size_t)(s & 1) * HBUF;
    _Float16* hnext = h16 + (size_t)((s + 1) & 1) * HBUF;
    step_fused<<<dim3(H_ / 32, B_ / 128, 2), 256, 0, stream>>>(
        hprev, hnext, whhH, xpH, len, c32, h32, s);
  }

  final_fc<<<1, 256, 0, stream>>>(h32, fc_w, fc_b, out);
}
